// BiLSTM_CRF_6244882448779
// MI455X (gfx1250) — matrix operand order repaired and checked by reading, not compiled, not run on hardware
//
// BiLSTM + CRF Viterbi for MI455X (gfx1250), wave32 + WMMA bf16 +
// double-buffered async global->LDS tile staging (ASYNCcnt pipeline).
//
//  - Pre-pack all weights into WMMA B-operand tiles (per-lane contiguous 32B).
//  - Big input-projection GEMM (bf16 WMMA, f32 accum) for both directions.
//  - 512 sequential steps: fused fwd+bwd recurrent GEMM (blockIdx.z = dir),
//    then a pointwise LSTM gate/state update kernel.
//  - Emission GEMM (K=1024 split across hf/hb buffers, N padded 32->64).
//  - CRF Viterbi: one wave32 per batch, transitions in LDS, serial backtrace.

#include <hip/hip_runtime.h>
#include <hip/hip_bf16.h>
#include <stdint.h>

#define B_  128
#define T_  512
#define E_  256
#define H_  512
#define G4  (4 * H_)   // 2048 gate columns
#define NT  32         // tags
#define TAG_START 30
#define TAG_END   31

typedef __attribute__((ext_vector_type(16))) __bf16   v16bf;
typedef __attribute__((ext_vector_type(8)))  float    v8f;
typedef __attribute__((ext_vector_type(4)))  uint32_t v4u;   // trivial 16B vector (union-safe)
typedef __attribute__((ext_vector_type(4)))  int      v4i;   // matches async-LDS builtin params

#define AS1 __attribute__((address_space(1)))
#define AS3 __attribute__((address_space(3)))

#if defined(__has_builtin)
#  if __has_builtin(__builtin_amdgcn_global_load_async_to_lds_b128)
#    define USE_ASYNC_LDS 1
#  endif
#endif
#ifndef USE_ASYNC_LDS
#  define USE_ASYNC_LDS 0
#endif

union AFrag { v4u q[2]; v16bf v; };

__device__ __forceinline__ float sigmoidf_(float x) { return 1.0f / (1.0f + __expf(-x)); }

__device__ __forceinline__ void wait_async0()
{
#if USE_ASYNC_LDS
#  if __has_builtin(__builtin_amdgcn_s_wait_asynccnt)
    __builtin_amdgcn_s_wait_asynccnt(0);
#  else
    asm volatile("s_wait_asynccnt 0x0" ::: "memory");
#  endif
#endif
}

// ---------------------------------------------------------------------------
// Pack W (N x K, row-major f32) into WMMA B-operand tiles, bf16.
// Layout: [kt][nt][lane(32)][e(16)]; lane<16: col n = nt*16+lane, K = kt*32+e
//         lane>=16: col n = nt*16+(lane-16), K = kt*32+16+e
// => each lane's fragment is 16 contiguous bf16 (32B) -> two b128 loads.
// ---------------------------------------------------------------------------
__global__ void pack_b_kernel(const float* __restrict__ W, int N, int K,
                              __bf16* __restrict__ out, int ntilesPad)
{
    int total = (K >> 5) * ntilesPad * 32 * 16;
    int idx = blockIdx.x * 256 + threadIdx.x;
    if (idx >= total) return;
    int e    = idx & 15;
    int lane = (idx >> 4) & 31;
    int t    = idx >> 9;
    int nt   = t % ntilesPad;
    int kt   = t / ntilesPad;
    int n = nt * 16 + (lane & 15);
    int k = kt * 32 + ((lane < 16) ? 0 : 16) + e;
    float v = (n < N && k < K) ? W[(size_t)n * K + k] : 0.0f;
    out[idx] = (__bf16)v;
}

__global__ void f32_to_bf16_kernel(const float* __restrict__ in,
                                   __bf16* __restrict__ out, int n)
{
    int i = blockIdx.x * 256 + threadIdx.x;
    if (i < n) out[i] = (__bf16)in[i];
}

__global__ void init_state_kernel(float* __restrict__ h, float* __restrict__ c,
                                  __bf16* __restrict__ hb, int n)
{
    int i = blockIdx.x * 256 + threadIdx.x;
    if (i < n) { h[i] = 0.0f; c[i] = 0.0f; hb[i] = (__bf16)0.0f; }
}

// ---------------------------------------------------------------------------
// 64x64 workgroup GEMM tile, 4 waves (128 threads), each wave 32x32 = 2x2 WMMA.
// A: bf16 row-major (dual source split at ksplit, for the hf|hb concat case).
// B: pre-packed tiles (see pack_b_kernel). C: f32 row-major (+optional bias).
// A tiles double-buffered in LDS; staged with GLOBAL_LOAD_ASYNC_TO_LDS_B128
// (ASYNCcnt) when available, so the k+1 tile DMA overlaps the k-tile WMMAs.
// Padded LDS row stride (80B) -> conflict-free ds_load_b128 fragment reads
// giving lane half h k = 16h..16h+15, the same order as the packed B tiles.
// ---------------------------------------------------------------------------
__device__ __forceinline__ void gemm64x64(
    const __bf16* __restrict__ A1, int lda1, int ksplit,
    const __bf16* __restrict__ A2, int lda2,
    const v4u*    __restrict__ Bpack, int ntilesPad,
    float*        __restrict__ C, int ldc,
    const float*  __restrict__ biasC,
    int mBlock, int nBlock, int Nreal, int K)
{
    __shared__ uint32_t lds[2 * 64 * 20]; // 2 bufs x 64 rows x (16 dw + 4 pad) = 10 KB
    const int ldsBufDw = 64 * 20;

    const int tid   = threadIdx.x;      // 0..127
    const int lane  = tid & 31;
    const int wave  = tid >> 5;         // 0..3
    const int waveM = wave >> 1;        // 0..1
    const int waveN = wave & 1;         // 0..1
    const int l16   = lane & 15;
    const int hi    = (lane >> 4) & 1;  // lane half

    v8f acc[2][2] = {};

    const int srow  = tid >> 1;         // staging row 0..63
    const int shalf = tid & 1;          // staging half-row (32B)

    // stage one 64x32 bf16 A tile (this thread's 32B slice) into LDS buf
    auto issueTile = [&](int k0, int bufIdx) {
        const __bf16* srcRow = (k0 < ksplit)
            ? (A1 + (size_t)(mBlock + srow) * lda1 + k0)
            : (A2 + (size_t)(mBlock + srow) * lda2 + (k0 - ksplit));
        const __bf16* gp = srcRow + shalf * 16;                 // 32B slice
        uint32_t* dp = lds + bufIdx * ldsBufDw + srow * 20 + shalf * 8;
#if USE_ASYNC_LDS
        v4i* gsrc = (v4i*)const_cast<__bf16*>(gp);
        __builtin_amdgcn_global_load_async_to_lds_b128(
            (AS1 v4i*)gsrc, (AS3 v4i*)dp, 0,  0);
        __builtin_amdgcn_global_load_async_to_lds_b128(
            (AS1 v4i*)gsrc, (AS3 v4i*)dp, 16, 0);
#else
        const v4u* g = (const v4u*)gp;
        v4u v0 = g[0];
        v4u v1 = g[1];
        *(v4u*)(dp)     = v0;
        *(v4u*)(dp + 4) = v1;
#endif
    };

    const int nk = K >> 5;
    issueTile(0, 0);

    for (int kt = 0; kt < nk; ++kt) {
        wait_async0();          // my slice of tile kt has landed in LDS
        __syncthreads();        // everyone's slice has landed

        if (kt + 1 < nk)        // background DMA of tile kt+1 overlaps compute
            issueTile((kt + 1) * 32, (kt + 1) & 1);

        const uint32_t* cur = lds + (kt & 1) * ldsBufDw;

        AFrag a[2], b[2];
        #pragma unroll
        for (int nt_ = 0; nt_ < 2; ++nt_) {
            int nt = (nBlock >> 4) + waveN * 2 + nt_;
            const v4u* bp = Bpack + (((size_t)kt * ntilesPad + nt) * 32 + lane) * 2;
            b[nt_].q[0] = bp[0];
            b[nt_].q[1] = bp[1];
        }
        #pragma unroll
        for (int mt = 0; mt < 2; ++mt) {
            int row = waveM * 32 + mt * 16 + l16;
            const char* base = (const char*)cur + row * 80 + hi * 32;
            a[mt].q[0] = *(const v4u*)(base);
            a[mt].q[1] = *(const v4u*)(base + 16);
        }
        #pragma unroll
        for (int mt = 0; mt < 2; ++mt)
            #pragma unroll
            for (int nt_ = 0; nt_ < 2; ++nt_)
                acc[mt][nt_] = __builtin_amdgcn_wmma_f32_16x16x32_bf16(
                    false, a[mt].v, false, b[nt_].v,
                    (short)0, acc[mt][nt_], false, false);

        __syncthreads();        // all reads of buf kt&1 done before it is reused
    }

    // ---- epilogue: C layout — VGPR r, lanes 0-15 -> M=r, lanes 16-31 -> M=8+r
    #pragma unroll
    for (int mt = 0; mt < 2; ++mt) {
        #pragma unroll
        for (int nt_ = 0; nt_ < 2; ++nt_) {
            int col = nBlock + waveN * 32 + nt_ * 16 + l16;
            if (col < Nreal) {
                float bv = biasC ? biasC[col] : 0.0f;
                #pragma unroll
                for (int r = 0; r < 8; ++r) {
                    int row = mBlock + waveM * 32 + mt * 16 + hi * 8 + r;
                    C[(size_t)row * ldc + col] = acc[mt][nt_][r] + bv;
                }
            }
        }
    }
}

__global__ void __launch_bounds__(128) gemm_generic_kernel(
    const __bf16* A1, int lda1, int ksplit, const __bf16* A2, int lda2,
    const v4u* Bpack, int ntilesPad, float* C, int ldc,
    const float* biasC, int Nreal, int K)
{
    gemm64x64(A1, lda1, ksplit, A2, lda2, Bpack, ntilesPad, C, ldc, biasC,
              blockIdx.x * 64, blockIdx.y * 64, Nreal, K);
}

// Fused fwd+bwd recurrent GEMM: Rproj[dir] = h_bf16[dir] @ Whh[dir]^T
__global__ void __launch_bounds__(128) recurrent_gemm_kernel(
    const __bf16* __restrict__ hbf, const v4u* __restrict__ WhhPack,
    float* __restrict__ Rproj)
{
    int dir = blockIdx.z;
    const __bf16* A  = hbf + (size_t)dir * B_ * H_;
    const v4u*    Bp = WhhPack + (size_t)dir * (H_ / 32) * (G4 / 16) * 32 * 2;
    float*        Cc = Rproj + (size_t)dir * B_ * G4;
    gemm64x64(A, H_, 1 << 30, A, H_, Bp, G4 / 16, Cc, G4, nullptr,
              blockIdx.x * 64, blockIdx.y * 64, G4, H_);
}

// ---------------------------------------------------------------------------
// Pointwise LSTM gate/state update for step s (fwd at t=s, bwd at t=T-1-s).
// ---------------------------------------------------------------------------
__global__ void __launch_bounds__(256) lstm_pointwise_kernel(
    int s,
    const float* __restrict__ Xproj_f, const float* __restrict__ Xproj_b,
    const float* __restrict__ Rproj,
    const float* __restrict__ bias_f, const float* __restrict__ bias_b,
    const float* __restrict__ masks,
    float* __restrict__ hS, float* __restrict__ cS, __bf16* __restrict__ hbf,
    __bf16* __restrict__ hs_f, __bf16* __restrict__ hs_b)
{
    int idx = blockIdx.x * 256 + threadIdx.x;
    if (idx >= 2 * B_ * H_) return;
    int dir = idx / (B_ * H_);
    int rem = idx - dir * (B_ * H_);
    int b   = rem / H_;
    int j   = rem - b * H_;
    int t   = dir ? (T_ - 1 - s) : s;

    const float* Xp   = dir ? Xproj_b : Xproj_f;
    const float* bias = dir ? bias_b  : bias_f;
    size_t gbase = ((size_t)b * T_ + t) * G4;
    size_t rbase = (size_t)dir * B_ * G4 + (size_t)b * G4;

    float ig = Xp[gbase + j         ] + Rproj[rbase + j         ] + bias[j         ];
    float fg = Xp[gbase + j +   H_  ] + Rproj[rbase + j +   H_  ] + bias[j +   H_  ];
    float gg = Xp[gbase + j + 2*H_  ] + Rproj[rbase + j + 2*H_  ] + bias[j + 2*H_  ];
    float og = Xp[gbase + j + 3*H_  ] + Rproj[rbase + j + 3*H_  ] + bias[j + 3*H_  ];

    float i_ = sigmoidf_(ig), f_ = sigmoidf_(fg), o_ = sigmoidf_(og);
    float g_ = tanhf(gg);

    size_t sidx  = (size_t)dir * B_ * H_ + rem;
    float  c_old = cS[sidx], h_old = hS[sidx];
    float  m     = masks[(size_t)b * T_ + t];
    float  c_new = f_ * c_old + i_ * g_;
    float  h_new = o_ * tanhf(c_new);
    float  h = m * h_new + (1.0f - m) * h_old;
    float  c = m * c_new + (1.0f - m) * c_old;

    hS[sidx] = h; cS[sidx] = c; hbf[sidx] = (__bf16)h;
    __bf16* hs = dir ? hs_b : hs_f;
    hs[((size_t)b * T_ + t) * H_ + j] = (__bf16)h;
}

// ---------------------------------------------------------------------------
// CRF Viterbi: one wave32 per batch. Transitions cached in LDS; double-
// buffered alphas; lane 0 does the serial argmax + backtrace.
// ---------------------------------------------------------------------------
__global__ void __launch_bounds__(32) viterbi_kernel(
    const float* __restrict__ emis,   // [b*T + t][32]
    const float* __restrict__ masks,  // [b][t]
    const float* __restrict__ trans,  // [32][32]
    unsigned char* __restrict__ bps,  // [b][T-1][32]
    float* __restrict__ out)          // [0..B) scores, [B + b*T + t] tags
{
    int b = blockIdx.x, j = threadIdx.x;
    __shared__ float sT[NT * NT];
    __shared__ float sA[2][NT];
    __shared__ float sFin[NT];

    for (int i = 0; i < NT; ++i) sT[i * NT + j] = trans[i * NT + j];
    float alpha = emis[((size_t)b * T_) * NT + j] + trans[TAG_START * NT + j];
    sA[0][j] = alpha;
    __syncthreads();

    int buf = 0;
    for (int t = 1; t < T_; ++t) {
        float m = masks[(size_t)b * T_ + t];
        float best = -3.0e38f; int bi = 0;
        #pragma unroll
        for (int i = 0; i < NT; ++i) {
            float sc = sA[buf][i] + sT[i * NT + j];
            if (sc > best) { best = sc; bi = i; }
        }
        float scn = best + emis[((size_t)b * T_ + t) * NT + j];
        alpha = m * scn + (1.0f - m) * alpha;
        bps[((size_t)b * (T_ - 1) + (t - 1)) * NT + j] = (unsigned char)bi;
        buf ^= 1;
        sA[buf][j] = alpha;
        __syncthreads();
    }

    sFin[j] = alpha + sT[j * NT + TAG_END];
    __syncthreads();

    if (j == 0) {
        float best = sFin[0]; int bi = 0;
        for (int i = 1; i < NT; ++i) if (sFin[i] > best) { best = sFin[i]; bi = i; }
        out[b] = best;
        float* tags = out + B_ + (size_t)b * T_;
        int cur = bi;
        tags[T_ - 1] = (masks[(size_t)b * T_ + T_ - 1] > 0.0f) ? (float)cur : -1.0f;
        for (int t = T_ - 2; t >= 0; --t) {
            if (masks[(size_t)b * T_ + t + 1] > 0.0f)
                cur = (int)bps[((size_t)b * (T_ - 1) + t) * NT + cur];
            tags[t] = (masks[(size_t)b * T_ + t] > 0.0f) ? (float)cur : -1.0f;
        }
    }
}

// ---------------------------------------------------------------------------
extern "C" void kernel_launch(void* const* d_in, const int* in_sizes, int n_in,
                              void* d_out, int out_size, void* d_ws, size_t ws_size,
                              hipStream_t stream)
{
    const float* X      = (const float*)d_in[0];
    const float* masks  = (const float*)d_in[1];
    // d_in[2] = length (int32), unused — masks suffice
    const float* W_ih_f = (const float*)d_in[3];
    const float* W_hh_f = (const float*)d_in[4];
    const float* b_f    = (const float*)d_in[5];
    const float* W_ih_b = (const float*)d_in[6];
    const float* W_hh_b = (const float*)d_in[7];
    const float* b_b    = (const float*)d_in[8];
    const float* W_out  = (const float*)d_in[9];
    const float* b_out  = (const float*)d_in[10];
    const float* trans  = (const float*)d_in[11];

    // ---- carve workspace ----
    char* p = (char*)d_ws;
    auto carve = [&](size_t bytes) -> char* {
        char* r = p; p += (bytes + 255) & ~(size_t)255; return r;
    };
    const size_t wihPerDir = (size_t)(E_ / 32) * (G4 / 16) * 32 * 16;   // bf16 elems
    const size_t whhPerDir = (size_t)(H_ / 32) * (G4 / 16) * 32 * 16;
    const size_t woutElems = (size_t)(2 * H_ / 32) * 4 * 32 * 16;       // N padded to 64

    __bf16* Xbf     = (__bf16*)carve((size_t)B_ * T_ * E_ * 2);
    float*  Xproj_f = (float*) carve((size_t)B_ * T_ * G4 * 4);
    float*  Xproj_b = (float*) carve((size_t)B_ * T_ * G4 * 4);
    float*  Rproj   = (float*) carve((size_t)2 * B_ * G4 * 4);
    __bf16* WihP_f  = (__bf16*)carve(wihPerDir * 2);
    __bf16* WihP_b  = (__bf16*)carve(wihPerDir * 2);
    __bf16* WhhP    = (__bf16*)carve(whhPerDir * 2 * 2);   // [dir]
    __bf16* WoutP   = (__bf16*)carve(woutElems * 2);
    float*  hS      = (float*) carve((size_t)2 * B_ * H_ * 4);
    float*  cS      = (float*) carve((size_t)2 * B_ * H_ * 4);
    __bf16* hbf     = (__bf16*)carve((size_t)2 * B_ * H_ * 2);
    __bf16* hs_f    = (__bf16*)carve((size_t)B_ * T_ * H_ * 2);
    __bf16* hs_b    = (__bf16*)carve((size_t)B_ * T_ * H_ * 2);
    float*  emis    = (float*) carve((size_t)B_ * T_ * NT * 4);
    unsigned char* bps = (unsigned char*)carve((size_t)B_ * (T_ - 1) * NT);

    // ---- 1. pack weights into WMMA B-tiles (bf16) ----
    {
        int tot = (int)wihPerDir;
        pack_b_kernel<<<(tot + 255) / 256, 256, 0, stream>>>(W_ih_f, G4, E_, WihP_f, G4 / 16);
        pack_b_kernel<<<(tot + 255) / 256, 256, 0, stream>>>(W_ih_b, G4, E_, WihP_b, G4 / 16);
        tot = (int)whhPerDir;
        pack_b_kernel<<<(tot + 255) / 256, 256, 0, stream>>>(W_hh_f, G4, H_, WhhP, G4 / 16);
        pack_b_kernel<<<(tot + 255) / 256, 256, 0, stream>>>(W_hh_b, G4, H_, WhhP + whhPerDir, G4 / 16);
        tot = (int)woutElems;
        pack_b_kernel<<<(tot + 255) / 256, 256, 0, stream>>>(W_out, NT, 2 * H_, WoutP, 4);
    }

    // ---- 2. X -> bf16 ; init states ----
    {
        int n = B_ * T_ * E_;
        f32_to_bf16_kernel<<<(n + 255) / 256, 256, 0, stream>>>(X, Xbf, n);
        int m = 2 * B_ * H_;
        init_state_kernel<<<(m + 255) / 256, 256, 0, stream>>>(hS, cS, hbf, m);
    }

    // ---- 3. input projections (big WMMA GEMMs, no bias — folded in pointwise) ----
    {
        dim3 g((B_ * T_) / 64, G4 / 64);
        gemm_generic_kernel<<<g, 128, 0, stream>>>(Xbf, E_, 1 << 30, Xbf, E_,
            (const v4u*)WihP_f, G4 / 16, Xproj_f, G4, nullptr, G4, E_);
        gemm_generic_kernel<<<g, 128, 0, stream>>>(Xbf, E_, 1 << 30, Xbf, E_,
            (const v4u*)WihP_b, G4 / 16, Xproj_b, G4, nullptr, G4, E_);
    }

    // ---- 4. sequential scan: fused fwd+bwd recurrent GEMM + pointwise ----
    {
        dim3 gR(B_ / 64, G4 / 64, 2);
        int  nPW = 2 * B_ * H_;
        for (int s = 0; s < T_; ++s) {
            recurrent_gemm_kernel<<<gR, 128, 0, stream>>>(hbf, (const v4u*)WhhP, Rproj);
            lstm_pointwise_kernel<<<(nPW + 255) / 256, 256, 0, stream>>>(
                s, Xproj_f, Xproj_b, Rproj, b_f, b_b, masks, hS, cS, hbf, hs_f, hs_b);
        }
    }

    // ---- 5. emission GEMM: concat(hf,hb) @ W_out^T + b_out (N padded to 64) ----
    {
        dim3 g((B_ * T_) / 64, 1);
        gemm_generic_kernel<<<g, 128, 0, stream>>>(hs_f, H_, H_, hs_b, H_,
            (const v4u*)WoutP, 4, emis, NT, b_out, NT, 2 * H_);
    }

    // ---- 6. CRF Viterbi + backtrace ----
    viterbi_kernel<<<B_, 32, 0, stream>>>(emis, masks, trans, bps, (float*)d_out);
}
